// TrackLoss_41626823032979
// MI455X (gfx1250) — compile-verified, matrix operand order checked by reading, not run on hardware
//
#include <hip/hip_runtime.h>
#include <math.h>

typedef __attribute__((ext_vector_type(2)))  float    v2f;
typedef __attribute__((ext_vector_type(8)))  float    v8f;
typedef __attribute__((ext_vector_type(16))) _Float16 v16h;

#define TRAJ_NUM 25
#define TPTS     10          // EVAL_POINTS//3 mid-segment time samples
#define NUM_LOS  10
#define VFOV_F   0.34906585f // deg2rad(20)

// ---------------------------------------------------------------------------
// Kernel 1: pos/vel via WMMA.
// pos[b,t,c] = sum_j (sum_i t^i L[i][j]) * d[b,c,j]   (TP*L folded into A)
// vel[b,t,c] = sum_j (sum_{k=1..5} k t^{k-1} L[k][j]) * d[b,c,j]
// GEMM: A [16x6] (rows=t, 10 valid), B [6 x 16cols], cols = (b,c) pairs.
// ---------------------------------------------------------------------------
__global__ __launch_bounds__(256)
void poly_eval_wmma(const float* __restrict__ Df, const float* __restrict__ Dp,
                    const float* __restrict__ L,  float* __restrict__ pos_ws,
                    float* __restrict__ vel_ws, int B)
{
    __shared__ float Ls[36];
    const int tid = threadIdx.x;
    if (tid < 36) Ls[tid] = L[tid];
    __syncthreads();

    const int lane  = tid & 31;
    const int wave  = tid >> 5;
    const int tile  = blockIdx.x * 8 + wave;
    const int ncols = B * 3;
    if (tile * 16 >= ncols) return;
    const int col = tile * 16 + (lane & 15);

    // ---- A operands: folded time-power x L rows (per lane: row m = lane&15)
    const int   m  = lane & 15;
    const float tm = 0.66f + (0.66f / 9.0f) * (float)m;   // linspace(0.66,1.32,10)
    float TPL[6], TVL[6];
    {
        float powi[6];
        powi[0] = 1.0f;
        #pragma unroll
        for (int i = 1; i < 6; ++i) powi[i] = powi[i-1] * tm;
        #pragma unroll
        for (int j = 0; j < 6; ++j) {
            float p = 0.0f, v = 0.0f;
            #pragma unroll
            for (int i = 0; i < 6; ++i) p += powi[i] * Ls[i*6 + j];
            #pragma unroll
            for (int k = 1; k < 6; ++k) v += (float)k * powi[k-1] * Ls[k*6 + j];
            TPL[j] = p; TVL[j] = v;
        }
    }
    if (m >= TPTS) {            // rows 10..15 are zero padding
        #pragma unroll
        for (int j = 0; j < 6; ++j) { TPL[j] = 0.0f; TVL[j] = 0.0f; }
    }

    // ---- B operand: raw derivative params d = [Df | Dp] for this column
    const int  b        = col / 3;
    const int  c        = col - 3 * b;
    const bool colvalid = (col < ncols);
    float dcol[6];
    if (colvalid) {
        const float* df = Df + (size_t)b * 9 + (size_t)c * 3;
        const float* dp = Dp + (size_t)b * 9 + (size_t)c * 3;
        dcol[0] = df[0]; dcol[1] = df[1]; dcol[2] = df[2];
        dcol[3] = dp[0]; dcol[4] = dp[1]; dcol[5] = dp[2];
    } else {
        #pragma unroll
        for (int j = 0; j < 6; ++j) dcol[j] = 0.0f;
    }

    v8f cpos = {}; v8f cvel = {};
    const bool hi = (lane >= 16);

#if __has_builtin(__builtin_amdgcn_wmma_f32_16x16x4_f32)
    // Exact f32 path: K=6 in two K=4 chunks (cols 6,7 zero-padded).
    // A 16x4 layout: lanes0-15 VGPR{0,1}=K{0,1}; lanes16-31 VGPR{0,1}=K{2,3}.
    v2f apos0, apos1, avel0, avel1, b0, b1;
    apos0.x = hi ? TPL[2] : TPL[0];  apos0.y = hi ? TPL[3] : TPL[1];
    avel0.x = hi ? TVL[2] : TVL[0];  avel0.y = hi ? TVL[3] : TVL[1];
    apos1.x = hi ? 0.0f   : TPL[4];  apos1.y = hi ? 0.0f   : TPL[5];
    avel1.x = hi ? 0.0f   : TVL[4];  avel1.y = hi ? 0.0f   : TVL[5];
    b0.x    = hi ? dcol[2] : dcol[0]; b0.y   = hi ? dcol[3] : dcol[1];
    b1.x    = hi ? 0.0f    : dcol[4]; b1.y   = hi ? 0.0f    : dcol[5];
    cpos = __builtin_amdgcn_wmma_f32_16x16x4_f32(false, apos0, false, b0, (short)0, cpos, false, false);
    cpos = __builtin_amdgcn_wmma_f32_16x16x4_f32(false, apos1, false, b1, (short)0, cpos, false, false);
    cvel = __builtin_amdgcn_wmma_f32_16x16x4_f32(false, avel0, false, b0, (short)0, cvel, false, false);
    cvel = __builtin_amdgcn_wmma_f32_16x16x4_f32(false, avel1, false, b1, (short)0, cvel, false, false);
#else
    // Codegen-confirmed fallback: f16 16x16x32 (K=0..5 nonzero, rest zero;
    // lanes 16-31 carry K>=8 => all zero).
    v16h ap = {}; v16h av = {}; v16h bb = {};
    if (!hi) {
        #pragma unroll
        for (int j = 0; j < 6; ++j) {
            ap[j] = (_Float16)TPL[j];
            av[j] = (_Float16)TVL[j];
            bb[j] = (_Float16)dcol[j];
        }
    }
    cpos = __builtin_amdgcn_wmma_f32_16x16x32_f16(false, ap, false, bb, (short)0, cpos, false, false);
    cvel = __builtin_amdgcn_wmma_f32_16x16x32_f16(false, av, false, bb, (short)0, cvel, false, false);
#endif

    // D layout: VGPR r -> t=r (lanes0-15) / t=r+8 (lanes16-31); store t<10.
    if (colvalid) {
        #pragma unroll
        for (int r = 0; r < 8; ++r) {
            int t = hi ? (r + 8) : r;
            if (t < TPTS) {
                size_t off = ((size_t)b * TPTS + t) * 3 + c;
                pos_ws[off] = cpos[r];
                vel_ws[off] = cvel[r];
            }
        }
    }
}

// ---------------------------------------------------------------------------
// Kernel 2: line-of-sight trilinear SDF sampling + occlusion + pitch penalty.
// One thread per (trajectory b, time t); gather-bound. Deterministic LDS
// reduction of the 10 time points per trajectory (no float atomics).
// ---------------------------------------------------------------------------
__global__ __launch_bounds__(320)
void sample_reduce(const float* __restrict__ pos_ws, const float* __restrict__ vel_ws,
                   const float* __restrict__ goal,   const float* __restrict__ sdf,
                   const float* __restrict__ min_bounds, const int* __restrict__ map_id,
                   float* __restrict__ out, int B, int G)
{
    __shared__ float red[320];
    const int tid    = threadIdx.x;
    const int localB = tid / TPTS;          // 0..31
    const int t      = tid - localB * TPTS; // 0..9
    const int b      = blockIdx.x * 32 + localB;

    float contrib = 0.0f;
    if (b < B) {
        const int o = b / TRAJ_NUM;
        const int mid = map_id[o];
        const float* __restrict__ sm = sdf + (size_t)mid * G * G * G;
        const float mbx = min_bounds[mid*3+0];
        const float mby = min_bounds[mid*3+1];
        const float mbz = min_bounds[mid*3+2];

        const size_t po = ((size_t)b * TPTS + t) * 3;
        const float px = pos_ws[po], py = pos_ws[po+1], pz = pos_ws[po+2];
        const float vx = vel_ws[po], vy = vel_ws[po+1], vz = vel_ws[po+2];

        const size_t go = (size_t)o * TRAJ_NUM * 3;   // goal of traj 0 in group
        const float gx = goal[go], gy = goal[go+1], gz = goal[go+2];
        const float dx = gx - px, dy = gy - py, dz = gz - pz;

        const float hib = (float)G - 1.5f;
        float acc = 0.0f;
        #pragma unroll
        for (int s = 0; s < NUM_LOS; ++s) {
            const float tt = (float)s * (1.0f / 9.0f);
            const float wx = (px + tt*dx - mbx) * 5.0f;   // 1/VOXEL
            const float wy = (py + tt*dy - mby) * 5.0f;
            const float wz = (pz + tt*dz - mbz) * 5.0f;
            const bool valid = (wx > 0.5f) && (wx < hib) && (wy > 0.5f) &&
                               (wy < hib) && (wz > 0.5f) && (wz < hib);
            float cost = 0.0f;
            if (valid) {
                const int x0 = (int)fminf(fmaxf(floorf(wx), 0.0f), (float)(G-2));
                const int y0 = (int)fminf(fmaxf(floorf(wy), 0.0f), (float)(G-2));
                const int z0 = (int)fminf(fmaxf(floorf(wz), 0.0f), (float)(G-2));
                const float fx = wx - (float)x0;
                const float fy = wy - (float)y0;
                const float fz = wz - (float)z0;
                const size_t base = ((size_t)z0 * G + y0) * G + x0;
                const size_t bz1  = base + (size_t)G * G;
                const float c000 = sm[base],     c001 = sm[base+1];
                const float c010 = sm[base+G],   c011 = sm[base+G+1];
                const float c100 = sm[bz1],      c101 = sm[bz1+1];
                const float c110 = sm[bz1+G],    c111 = sm[bz1+G+1];
                const float l00 = c000 + fx*(c001-c000);
                const float l01 = c010 + fx*(c011-c010);
                const float l10 = c100 + fx*(c101-c100);
                const float l11 = c110 + fx*(c111-c110);
                const float l0  = l00 + fy*(l01-l00);
                const float l1  = l10 + fy*(l11-l10);
                const float dist = l0 + fz*(l1-l0);
                cost = __expf(-(dist - 0.6f) * (1.0f/0.3f));
            }
            acc += fmaxf(0.2f - cost, 0.0f);   // invalid -> contributes 0.2
        }
        const float occ = acc * (1.0f / NUM_LOS);
        const float pg = atan2f(dz, sqrtf(dx*dx + dy*dy + 1e-6f));
        const float pv = atan2f(vz, sqrtf(vx*vx + vy*vy + 1e-6f));
        const float pen = fmaxf(fabsf(pg - pv) - VFOV_F, 0.0f);
        contrib = occ * 4.0f + pen;
    }
    red[tid] = contrib;
    __syncthreads();
    if (t == 0 && b < B) {
        float s = 0.0f;
        #pragma unroll
        for (int i = 0; i < TPTS; ++i) s += red[localB * TPTS + i];
        out[b] = s * 0.5f;   // mean over 10, times 5
    }
}

// ---------------------------------------------------------------------------
extern "C" void kernel_launch(void* const* d_in, const int* in_sizes, int n_in,
                              void* d_out, int out_size, void* d_ws, size_t ws_size,
                              hipStream_t stream)
{
    const float* Df     = (const float*)d_in[0];
    const float* Dp     = (const float*)d_in[1];
    const float* goal   = (const float*)d_in[2];
    const float* L      = (const float*)d_in[3];
    const float* sdf    = (const float*)d_in[4];
    const float* minb   = (const float*)d_in[5];
    const int*   map_id = (const int*)d_in[6];
    float*       out    = (float*)d_out;

    const int B = in_sizes[0] / 9;              // [B,3,3]
    const int M = in_sizes[5] / 3;              // [M,3]
    const long vox = (long)in_sizes[4] / (M > 0 ? M : 1);
    const int G = (int)llroundf(cbrtf((float)vox));   // 200

    float* pos_ws = (float*)d_ws;
    float* vel_ws = pos_ws + (size_t)B * TPTS * 3;

    const int ncols   = B * 3;
    const int tiles   = (ncols + 15) / 16;
    const int blocks1 = (tiles + 7) / 8;        // 8 waves (tiles) per block
    poly_eval_wmma<<<blocks1, 256, 0, stream>>>(Df, Dp, L, pos_ws, vel_ws, B);

    const int blocks2 = (B + 31) / 32;
    sample_reduce<<<blocks2, 320, 0, stream>>>(pos_ws, vel_ws, goal, sdf, minb,
                                               map_id, out, B, G);
}
